// MHFSpectralConv_22067541967407
// MI455X (gfx1250) — matrix operand order repaired and checked by reading, not compiled, not run on hardware
//
#include <hip/hip_runtime.h>
#include <hip/hip_bf16.h>
#include <math.h>

typedef __bf16 bf16_t;
typedef bf16_t v16bf __attribute__((ext_vector_type(16)));
typedef float  v8f   __attribute__((ext_vector_type(8)));

#define B_      16
#define C_IN_   128
#define C_OUT_  128
#define H_      8
#define MODES_  256
#define L_      16384
#define F2_     512            // 2*MODES (re || im)
#define TWO_PI_ 6.28318530717958647692f

#if defined(__has_builtin)
#if __has_builtin(__builtin_amdgcn_global_load_async_to_lds_b32) && \
    __has_builtin(__builtin_amdgcn_s_wait_asynccnt)
#define USE_ASYNC_LDS 1
#endif
#endif

typedef __attribute__((address_space(1))) int* gas_i32p;
typedef __attribute__((address_space(3))) int* las_i32p;

static __device__ __forceinline__ v8f wmma_bf16(v16bf a, v16bf b, v8f c) {
  // D = A(16x32 bf16) * B(32x16 bf16) + C(16x16 f32)
  return __builtin_amdgcn_wmma_f32_16x16x32_bf16(false, a, false, b, (short)0, c,
                                                 false, false);
}

// ---------------------------------------------------------------------------
// Basis init: fwdB[n][l], n<256: cos(2*pi*n*l/L)/L ; n>=256: -sin(...)/L
// (rfft norm='forward':  Re = (1/L) sum x cos,  Im = -(1/L) sum x sin)
// ---------------------------------------------------------------------------
__global__ void __launch_bounds__(256) k_fwd_basis(bf16_t* __restrict__ fwdB) {
  int idx = blockIdx.x * 256 + threadIdx.x;       // < 512 * 16384
  int n = idx >> 14;
  int l = idx & (L_ - 1);
  int f = (n < MODES_) ? n : (n - MODES_);
  int ph = (f * l) & (L_ - 1);                    // (f*l) mod L, L pow2
  float ang = (float)ph * (TWO_PI_ / (float)L_);
  float v = (n < MODES_) ? cosf(ang) : -sinf(ang);
  fwdB[idx] = (bf16_t)(v * (1.0f / (float)L_));
}

// invT[l][k]: k<256: w*cos(2*pi*k*l/L) ; k>=256: -w*sin(...) ; w = (f==0)?1:2
// (irfft norm='forward' + hermitian fold of truncated spectrum)
__global__ void __launch_bounds__(256) k_inv_basis(bf16_t* __restrict__ invT) {
  int idx = blockIdx.x * 256 + threadIdx.x;       // < 16384 * 512
  int l = idx >> 9;
  int k = idx & (F2_ - 1);
  int f = (k < MODES_) ? k : (k - MODES_);
  float w = (f == 0) ? 1.0f : 2.0f;
  int ph = (f * l) & (L_ - 1);
  float ang = (float)ph * (TWO_PI_ / (float)L_);
  float v = (k < MODES_) ? (w * cosf(ang)) : (-w * sinf(ang));
  invT[idx] = (bf16_t)v;
}

// ---------------------------------------------------------------------------
// Stage 1: truncated forward DFT as GEMM.
//   xf[m][n] = sum_l x[m][l] * fwdB[n][l],  M=2048, K=16384, N=512
// One wave -> 16x64 tile; all four B fragments fetched before the WMMAs so
// the loads clause together and waits stagger instead of stalling each WMMA.
// ---------------------------------------------------------------------------
__global__ void __launch_bounds__(256) k_dft(const float* __restrict__ x,
                                             const bf16_t* __restrict__ fwdB,
                                             float* __restrict__ xf) {
  int gw   = blockIdx.x * 8 + (threadIdx.x >> 5);   // 0..1023
  int mt   = gw >> 3;                               // 0..127
  int n0   = (gw & 7) * 64;                         // 0..448
  int lane = threadIdx.x & 31;
  int m    = mt * 16 + (lane & 15);
  int kb8  = (lane >> 4) * 8;
  int kb16 = (lane >> 4) * 16;
  int ncol = lane & 15;

  const float*  pA0 = x    + (size_t)m * L_;
  const bf16_t* pB0 = fwdB + (size_t)(n0 + ncol) * L_;

  v8f acc[4] = {};
#pragma unroll 2
  for (int k0 = 0; k0 < L_; k0 += 32) {
    const float* pA = pA0 + k0 + kb16;
    __builtin_prefetch(pA + 256, 0, 0);             // global_prefetch_b8
    v16bf a;
#pragma unroll
    for (int j = 0; j < 8; ++j) {                   // A-frag: contiguous K (16h + s)
      a[j]     = (bf16_t)pA[j];
      a[8 + j] = (bf16_t)pA[8 + j];
    }
    v16bf b[4];
#pragma unroll
    for (int t = 0; t < 4; ++t) {                   // fetch phase (one clause)
      const bf16_t* pB = pB0 + (size_t)t * 16 * L_ + k0 + kb16;
#pragma unroll
      for (int j = 0; j < 16; ++j) b[t][j] = pB[j];
    }
#pragma unroll
    for (int t = 0; t < 4; ++t)                     // compute phase
      acc[t] = wmma_bf16(a, b[t], acc[t]);
  }
  int mb = mt * 16 + (lane >> 4) * 8;
#pragma unroll
  for (int t = 0; t < 4; ++t)
#pragma unroll
    for (int r = 0; r < 8; ++r)
      xf[(size_t)(mb + r) * F2_ + n0 + t * 16 + ncol] = acc[t][r];
}

// ---------------------------------------------------------------------------
// Stage 2: per-(h,f) complex 16x16x16 head mix; K=16 zero-padded to 32.
//   of[b*128+h*16+o][f]     = Re:  XRe*WRe - XIm*WIm
//   of[b*128+h*16+o][256+f] = Im:  XRe*WIm + XIm*WRe
// ---------------------------------------------------------------------------
__global__ void __launch_bounds__(256) k_mix(const float* __restrict__ xf,
                                             const float* __restrict__ wre,
                                             const float* __restrict__ wim,
                                             float* __restrict__ of) {
  int gw   = blockIdx.x * 8 + (threadIdx.x >> 5);   // 0..2047
  int h    = gw >> 8;
  int f    = gw & 255;
  int lane = threadIdx.x & 31;
  int bb   = lane & 15;                             // A-frag row (batch)
  int oo   = lane & 15;                             // B-frag col (out chan)
  int kb8  = (lane >> 4) * 8;
  int kb16 = (lane >> 4) * 16;

  v16bf aRe, aIm;
#pragma unroll
  for (int j = 0; j < 8; ++j) {
    int i = kb8 + j;                                // input head channel
    size_t base = ((size_t)(bb * C_IN_) + h * 16 + i) * F2_ + f;
    aRe[j] = (bf16_t)xf[base];
    aIm[j] = (bf16_t)xf[base + MODES_];
    aRe[8 + j] = (bf16_t)0.0f;                      // K 16..31 padding
    aIm[8 + j] = (bf16_t)0.0f;
  }
  v16bf bRe, bIm, bImN;
#pragma unroll
  for (int j = 0; j < 16; ++j) {
    int k = (j < 8) ? (kb8 + j) : (16 + kb8 + j - 8);
    if (k < 16) {
      size_t wi = (((size_t)(h * 16 + k)) * 16 + oo) * MODES_ + f;
      float wr = wre[wi], wv = wim[wi];
      bRe[j]  = (bf16_t)wr;
      bIm[j]  = (bf16_t)wv;
      bImN[j] = (bf16_t)(-wv);                      // NEG only hits C, so negate here
    } else {
      bRe[j] = (bf16_t)0.0f; bIm[j] = (bf16_t)0.0f; bImN[j] = (bf16_t)0.0f;
    }
  }
  v8f accRe = {}, accIm = {};
  accRe = wmma_bf16(aRe, bRe,  accRe);
  accRe = wmma_bf16(aIm, bImN, accRe);
  accIm = wmma_bf16(aRe, bIm,  accIm);
  accIm = wmma_bf16(aIm, bRe,  accIm);

  int b0 = (lane >> 4) * 8;
#pragma unroll
  for (int r = 0; r < 8; ++r) {
    int b = b0 + r;
    size_t base = ((size_t)(b * C_OUT_) + h * 16 + oo) * F2_ + f;
    of[base]          = accRe[r];
    of[base + MODES_] = accIm[r];
  }
}

// ---------------------------------------------------------------------------
// Conv branch. x tile staged to LDS via GLOBAL_LOAD_ASYNC_TO_LDS (ASYNCcnt)
// when the toolchain exposes it; WMMA GEMM M=C_OUT(128), N=64, K=3*128=384.
// Writes d_out = (1-g) * (conv + conv_b).
// ---------------------------------------------------------------------------
#define XT_ROWS 66
__global__ void __launch_bounds__(256) k_conv(const float* __restrict__ x,
                                              const float* __restrict__ conv_w,
                                              const float* __restrict__ conv_b,
                                              const float* __restrict__ gate,
                                              float* __restrict__ out) {
  __shared__ float xTf[XT_ROWS * 128];              // [l_local 0..65][ci], f32
  int b  = blockIdx.x >> 8;                         // 16 batches
  int l0 = (blockIdx.x & 255) * 64;                 // 256 l-groups

#if defined(USE_ASYNC_LDS)
  for (int idx = threadIdx.x; idx < XT_ROWS * 128; idx += 256) {
    int row = idx >> 7;                             // l_local, halo of 1 each side
    int ci  = idx & 127;
    int l   = l0 - 1 + row;
    if (l >= 0 && l < L_) {
      float* gp = const_cast<float*>(x) + ((size_t)(b * C_IN_) + ci) * L_ + l;
      __builtin_amdgcn_global_load_async_to_lds_b32(
          (gas_i32p)(void*)gp,
          (las_i32p)(void*)&xTf[idx], 0, 0);
    } else {
      xTf[idx] = 0.0f;                              // zero halo directly
    }
  }
  __builtin_amdgcn_s_wait_asynccnt(0);              // my async copies landed
  __syncthreads();                                  // everyone's landed
#else
  for (int idx = threadIdx.x; idx < XT_ROWS * 128; idx += 256) {
    int row = idx >> 7;
    int ci  = idx & 127;
    int l   = l0 - 1 + row;
    xTf[idx] = (l >= 0 && l < L_) ? x[((size_t)(b * C_IN_) + ci) * L_ + l] : 0.0f;
  }
  __syncthreads();
#endif

  int wave = threadIdx.x >> 5;                      // 8 waves -> 8 co-tiles
  int lane = threadIdx.x & 31;
  int co0  = wave * 16;
  int kb8  = (lane >> 4) * 8;
  int kb16 = (lane >> 4) * 16;

  v8f acc[4] = {};
  for (int kt = 0; kt < 12; ++kt) {                 // K = 384, k = t*128+ci
    int k0 = kt * 32;
    v16bf a;                                        // conv_w fragment
    int ma = co0 + (lane & 15);
#pragma unroll
    for (int j = 0; j < 8; ++j) {
      int k  = k0 + kb8 + j;       int t  = k >> 7;  int ci  = k & 127;
      int k2 = k0 + 16 + kb8 + j;  int t2 = k2 >> 7; int ci2 = k2 & 127;
      a[j]     = (bf16_t)conv_w[((size_t)ma * 128 + ci) * 3 + t];
      a[8 + j] = (bf16_t)conv_w[((size_t)ma * 128 + ci2) * 3 + t2];
    }
#pragma unroll
    for (int s = 0; s < 4; ++s) {
      v16bf bf;
      int n = lane & 15;
#pragma unroll
      for (int j = 0; j < 16; ++j) {
        int k = k0 + ((j < 8) ? (kb8 + j) : (16 + kb8 + j - 8)); int t = k >> 7; int ci = k & 127;
        bf[j] = (bf16_t)xTf[(s * 16 + n + t) * 128 + ci];  // row = l_local + t
      }
      acc[s] = wmma_bf16(a, bf, acc[s]);
    }
  }

  float g  = 1.0f / (1.0f + expf(-gate[0]));
  float om = 1.0f - g;
  int nc = lane & 15;
  int mb = co0 + (lane >> 4) * 8;
#pragma unroll
  for (int s = 0; s < 4; ++s)
#pragma unroll
    for (int r = 0; r < 8; ++r) {
      int co = mb + r;
      int l  = l0 + s * 16 + nc;
      out[((size_t)(b * C_OUT_) + co) * L_ + l] = om * (acc[s][r] + conv_b[co]);
    }
}

// ---------------------------------------------------------------------------
// Inverse DFT GEMM + fused blend (runs AFTER k_conv).
//   out[m][l] += g * (sum_k of[m][k] * invT[l][k] + bias[m%128])
//   M=2048, K=512, N=16384. One wave -> 16x64 tile.
// ---------------------------------------------------------------------------
__global__ void __launch_bounds__(256) k_idft(const float* __restrict__ of,
                                              const bf16_t* __restrict__ invT,
                                              const float* __restrict__ bias,
                                              const float* __restrict__ gate,
                                              float* __restrict__ out) {
  int gw   = blockIdx.x * 8 + (threadIdx.x >> 5);   // 0..32767
  int mt   = gw >> 8;                               // 0..127
  int l0   = (gw & 255) * 64;                       // 0..16320
  int lane = threadIdx.x & 31;
  int m    = mt * 16 + (lane & 15);
  int kb8  = (lane >> 4) * 8;
  int kb16 = (lane >> 4) * 16;
  int ncol = lane & 15;

  const float*  pA0 = of   + (size_t)m * F2_;
  const bf16_t* pB0 = invT + (size_t)(l0 + ncol) * F2_;

  v8f acc[4] = {};
#pragma unroll
  for (int k0 = 0; k0 < F2_; k0 += 32) {
    const float* pA = pA0 + k0 + kb16;
    v16bf a;
#pragma unroll
    for (int j = 0; j < 8; ++j) {
      a[j]     = (bf16_t)pA[j];
      a[8 + j] = (bf16_t)pA[8 + j];
    }
    v16bf b[4];
#pragma unroll
    for (int t = 0; t < 4; ++t) {                   // fetch phase
      const bf16_t* pB = pB0 + (size_t)t * 16 * F2_ + k0 + kb16;
#pragma unroll
      for (int j = 0; j < 16; ++j) b[t][j] = pB[j];
    }
#pragma unroll
    for (int t = 0; t < 4; ++t)                     // compute phase
      acc[t] = wmma_bf16(a, b[t], acc[t]);
  }

  float g = 1.0f / (1.0f + expf(-gate[0]));
  int mb = mt * 16 + (lane >> 4) * 8;
#pragma unroll
  for (int t = 0; t < 4; ++t)
#pragma unroll
    for (int r = 0; r < 8; ++r) {
      int mm = mb + r;
      int co = mm & 127;
      size_t idx = (size_t)mm * L_ + l0 + t * 16 + ncol;
      out[idx] = out[idx] + g * (acc[t][r] + bias[co]);
    }
}

// ---------------------------------------------------------------------------
extern "C" void kernel_launch(void* const* d_in, const int* in_sizes, int n_in,
                              void* d_out, int out_size, void* d_ws, size_t ws_size,
                              hipStream_t stream) {
  const float* x      = (const float*)d_in[0];
  const float* wre    = (const float*)d_in[1];
  const float* wim    = (const float*)d_in[2];
  const float* bias   = (const float*)d_in[3];
  const float* conv_w = (const float*)d_in[4];
  const float* conv_b = (const float*)d_in[5];
  const float* gate   = (const float*)d_in[6];
  float* out = (float*)d_out;

  // Workspace layout (40 MiB total):
  //  [0,16M)   fwd basis bf16  [512][16384]
  //  [16,32M)  inv basis bf16  [16384][512]
  //  [32,36M)  xf f32 [2048][512]
  //  [36,40M)  of f32 [2048][512]
  char* ws = (char*)d_ws;
  bf16_t* fwdB = (bf16_t*)(ws);
  bf16_t* invT = (bf16_t*)(ws + (size_t)16 * 1024 * 1024);
  float*  xf   = (float*)(ws + (size_t)32 * 1024 * 1024);
  float*  of   = (float*)(ws + (size_t)36 * 1024 * 1024);

  k_fwd_basis<<<32768, 256, 0, stream>>>(fwdB);
  k_inv_basis<<<32768, 256, 0, stream>>>(invT);
  k_dft      <<<  128, 256, 0, stream>>>(x, fwdB, xf);
  k_mix      <<<  256, 256, 0, stream>>>(xf, wre, wim, of);
  k_conv     <<< 4096, 256, 0, stream>>>(x, conv_w, conv_b, gate, out);  // writes (1-g)*local
  k_idft     <<< 4096, 256, 0, stream>>>(of, invT, bias, gate, out);     // += g*(spectral+bias)
}
